// StrictDualCausalSelfAttention_4243427688651
// MI455X (gfx1250) — matrix operand order repaired and checked by reading, not compiled, not run on hardware
//
#include <hip/hip_runtime.h>

// ---------------------------------------------------------------------------
// StrictDualCausalSelfAttention for MI455X (gfx1250, wave32, WMMA)
// S=2048, B=2, E=1024, H=16, D=64
// All matmuls on v_wmma_f32_16x16x32_f16. Attention runs 16 heads of one
// query tile in one 16-wave workgroup: head-mean is a cross-wave LDS
// reduction (no global atomics). V staged via global_load_async_to_lds_b128.
// ---------------------------------------------------------------------------

typedef __attribute__((ext_vector_type(16))) _Float16     v16h;
typedef __attribute__((ext_vector_type(8)))  _Float16     v8h;
typedef __attribute__((ext_vector_type(8)))  float        v8f;
typedef __attribute__((ext_vector_type(4)))  unsigned int v4u;
typedef _Float16 half_t;

constexpr int S = 2048;
constexpr int B = 2;
constexpr int E = 1024;
constexpr int H = 16;
constexpr int D = 64;                 // E / H
constexpr float QK_SCALE = 0.125f;    // 1/sqrt(D), folded into Q projection
constexpr float NEG_BIG  = -1e30f;    // finite -inf sentinel (online-softmax safe)

__device__ __forceinline__ v8f wmma_f16(v16h a, v16h b, v8f c) {
  return __builtin_amdgcn_wmma_f32_16x16x32_f16(false, a, false, b, (short)0, c,
                                                false, false);
}

__device__ __forceinline__ v16h cat8(v8h lo, v8h hi) {
  v16h r;
#pragma unroll
  for (int i = 0; i < 8; ++i) { r[i] = lo[i]; r[i + 8] = hi[i]; }
  return r;
}

// gfx1250 async global->LDS copy (ASYNCcnt-tracked), per ISA 10.x / 15.18.3
__device__ __forceinline__ void async_ld_b128(unsigned int lds_off, const void* g) {
  asm volatile("global_load_async_to_lds_b128 %0, %1, off"
               :: "v"(lds_off), "v"((unsigned long long)(uintptr_t)g)
               : "memory");
}
__device__ __forceinline__ void wait_async0() {
  asm volatile("s_wait_asynccnt 0x0" ::: "memory");
}
__device__ __forceinline__ void wait_ds0() {
  asm volatile("s_wait_dscnt 0x0" ::: "memory");
}

// ---------------------------------------------------------------------------
// Kernel 1: fused Q/K/V projection. One wave computes a 16x64 tile
// (4 accumulators, A-fragment reused 4x). Q is pre-scaled by 1/sqrt(D).
// f16 results stored [B][H][S][D]; epilogue coalesced through LDS.
// grid = (S/16, E/64, B*3)
// ---------------------------------------------------------------------------
__global__ void __launch_bounds__(32)
qkv_proj_kernel(const float* __restrict__ x,
                const float* __restrict__ qw, const float* __restrict__ qb,
                const float* __restrict__ kw, const float* __restrict__ kb,
                const float* __restrict__ vw, const float* __restrict__ vb,
                half_t* __restrict__ qws, half_t* __restrict__ kws,
                half_t* __restrict__ vws)
{
  __shared__ __align__(16) half_t tile[16 * 64];

  const int lane = threadIdx.x & 31;
  const int c  = lane & 15;
  const int hh = lane >> 4;
  const int s0 = blockIdx.x * 16;
  const int n0 = blockIdx.y * 64;        // one head's full D range
  const int b  = blockIdx.z / 3;
  const int m  = blockIdx.z % 3;

  const float* w    = (m == 0) ? qw : (m == 1) ? kw : vw;
  const float* bias = (m == 0) ? qb : (m == 1) ? kb : vb;
  half_t*      outp = (m == 0) ? qws : (m == 1) ? kws : vws;

  const float* xrow = x + ((size_t)(s0 + c) * B + b) * E;
  const float* wcol = w + (size_t)(n0 + c) * E;

  v8f acc[4] = {};
  for (int kk = 0; kk < E; kk += 32) {
    const v8f xa0 = *(const v8f*)(xrow + kk + hh * 16);
    const v8f xa1 = *(const v8f*)(xrow + kk + hh * 16 + 8);
    v16h a;
#pragma unroll
    for (int i = 0; i < 8; ++i) { a[i] = (half_t)xa0[i]; a[i + 8] = (half_t)xa1[i]; }

#pragma unroll
    for (int nt = 0; nt < 4; ++nt) {
      const float* wr = wcol + (size_t)nt * 16 * E + kk + hh * 16;
      const v8f wb0 = *(const v8f*)wr;
      const v8f wb1 = *(const v8f*)(wr + 8);
      v16h bf;
#pragma unroll
      for (int i = 0; i < 8; ++i) { bf[i] = (half_t)wb0[i]; bf[i + 8] = (half_t)wb1[i]; }
      acc[nt] = wmma_f16(a, bf, acc[nt]);
    }
  }

  const float sc = (m == 0) ? QK_SCALE : 1.0f;
#pragma unroll
  for (int nt = 0; nt < 4; ++nt) {
#pragma unroll
    for (int v = 0; v < 8; ++v) {
      const int n = n0 + nt * 16 + c;
      tile[(v + 8 * hh) * 64 + nt * 16 + c] = (half_t)((acc[nt][v] + bias[n]) * sc);
    }
  }
  __syncthreads();

  const int r2  = lane & 15;
  const int seg = lane >> 4;
  const int ho  = n0 >> 6;
  half_t* dst = outp + (((size_t)b * H + ho) * S + (s0 + r2)) * D + seg * 32;
  const half_t* src = &tile[r2 * 64 + seg * 32];
#pragma unroll
  for (int t = 0; t < 4; ++t)
    *(v4u*)(dst + t * 8) = *(const v4u*)(src + t * 8);
}

// ---------------------------------------------------------------------------
// Kernel 2: attention. grid = (S/16, B), 16 waves per block (wave = head).
// Pass 1 per wave: WMMA score tiles + mask + online (m,l), shfl-xor merge.
// Pass 2 per 32-key chunk: probabilities -> pt[head] in LDS; cross-wave
// head-mean reduction with plain coalesced stores (no atomics anywhere);
// P @ V with async-LDS-staged V.
// ---------------------------------------------------------------------------
__global__ void __launch_bounds__(512)
attention_kernel(const half_t* __restrict__ qws, const half_t* __restrict__ kws,
                 const half_t* __restrict__ vws,
                 const unsigned char* __restrict__ cond,
                 const unsigned char* __restrict__ kpm,
                 float* __restrict__ attn_mean, half_t* __restrict__ ows)
{
  __shared__ __align__(16) half_t pt[H][16 * 32];    // per-head P tiles (16KB)
  __shared__ __align__(16) half_t vbuf[H][32 * 64];  // per-head V staging (64KB)
  __shared__ __align__(16) half_t otile[H][16 * 64]; // per-head output (32KB)

  const int lane = threadIdx.x & 31;
  const int wv   = threadIdx.x >> 5;   // wave id == head id
  const int c  = lane & 15;
  const int hh = lane >> 4;
  const int i0 = blockIdx.x * 16;
  const int h  = wv;
  const int b  = blockIdx.y;

  const half_t* Q = qws + (((size_t)b * H + h) * S) * D;
  const half_t* K = kws + (((size_t)b * H + h) * S) * D;
  const half_t* V = vws + (((size_t)b * H + h) * S) * D;
  const unsigned char* cb = cond + (size_t)b * S;
  const unsigned char* pb = kpm  + (size_t)b * S;

  // Q A-fragments (already scaled by 1/sqrt(D) in projection)
  const half_t* qrow = Q + (size_t)(i0 + c) * D;
  const v16h aq0 = cat8(*(const v8h*)(qrow + hh * 16),
                        *(const v8h*)(qrow + 8 + hh * 16));
  const v16h aq1 = cat8(*(const v8h*)(qrow + 32 + hh * 16),
                        *(const v8h*)(qrow + 40 + hh * 16));

  bool ci[8];
#pragma unroll
  for (int v = 0; v < 8; ++v) ci[v] = cb[i0 + v + 8 * hh] != 0;

  // ---- pass 1 (per wave, barrier-free): online row max / sum -------------
  float mrow[8], lrow[8];
#pragma unroll
  for (int v = 0; v < 8; ++v) { mrow[v] = NEG_BIG; lrow[v] = 0.0f; }

  for (int j0 = 0; j0 < S; j0 += 16) {
    if (j0 + 16 < S)
      __builtin_prefetch(K + (size_t)(j0 + 16 + c) * D, 0, 0);

    const half_t* krow = K + (size_t)(j0 + c) * D;
    const v16h bk0 = *(const v16h*)(krow + hh * 16);
    const v16h bk1 = *(const v16h*)(krow + 32 + hh * 16);
    v8f sc = {};
    sc = wmma_f16(aq0, bk0, sc);
    sc = wmma_f16(aq1, bk1, sc);

    const int  j   = j0 + c;
    const bool cj  = cb[j] != 0;
    const bool pad = pb[j] != 0;
#pragma unroll
    for (int v = 0; v < 8; ++v) {
      const int  i      = i0 + v + 8 * hh;
      const bool future = j > i;
      const bool masked = (future && !(ci[v] && cj)) || (ci[v] && !cj) || pad;
      const float s  = masked ? NEG_BIG : sc[v];
      const float mn = fmaxf(mrow[v], s);
      lrow[v] = lrow[v] * __expf(mrow[v] - mn) + __expf(s - mn);
      mrow[v] = mn;
    }
  }

#pragma unroll
  for (int off = 1; off < 16; off <<= 1) {
#pragma unroll
    for (int v = 0; v < 8; ++v) {
      const float mo = __shfl_xor(mrow[v], off, 32);
      const float lo = __shfl_xor(lrow[v], off, 32);
      const float mn = fmaxf(mrow[v], mo);
      lrow[v] = lrow[v] * __expf(mrow[v] - mn) + lo * __expf(mo - mn);
      mrow[v] = mn;
    }
  }
  float inv_l[8];
#pragma unroll
  for (int v = 0; v < 8; ++v) inv_l[v] = 1.0f / lrow[v];

  // ---- pass 2: probabilities, LDS head-mean reduction, P @ V -------------
  v8f acc[4] = {};

  for (int j0 = 0; j0 < S; j0 += 32) {
    // async-stage this head's V chunk; consumed after score/softmax work
    {
      wait_ds0();   // prior wave-local reads of vbuf[wv] retired
      const char* gbase = (const char*)(V + (size_t)j0 * D);
      const unsigned int lbase = (unsigned int)(uintptr_t)(&vbuf[wv][0]);
#pragma unroll
      for (int t = 0; t < 8; ++t) {
        const unsigned int off = (unsigned int)(t * 32 + lane) * 16;
        async_ld_b128(lbase + off, gbase + off);
      }
    }

#pragma unroll
    for (int t = 0; t < 2; ++t) {
      const int jb = j0 + t * 16;
      const half_t* krow = K + (size_t)(jb + c) * D;
      const v16h bk0 = *(const v16h*)(krow + hh * 16);
      const v16h bk1 = *(const v16h*)(krow + 32 + hh * 16);
      v8f sc = {};
      sc = wmma_f16(aq0, bk0, sc);
      sc = wmma_f16(aq1, bk1, sc);

      const int  j   = jb + c;
      const bool cj  = cb[j] != 0;
      const bool pad = pb[j] != 0;
#pragma unroll
      for (int v = 0; v < 8; ++v) {
        const int  i      = i0 + v + 8 * hh;
        const bool future = j > i;
        const bool masked = (future && !(ci[v] && cj)) || (ci[v] && !cj) || pad;
        const float p = masked ? 0.0f : __expf(sc[v] - mrow[v]) * inv_l[v];
        pt[wv][(v + 8 * hh) * 32 + t * 16 + c] = (half_t)p;
      }
    }
    __syncthreads();   // all heads' pt tiles for this chunk are visible

    // cross-wave head-mean: wave wv owns query row i0+wv, lanes own columns
    {
      float sum = 0.0f;
#pragma unroll
      for (int w2 = 0; w2 < H; ++w2)
        sum += (float)pt[w2][wv * 32 + lane];
      attn_mean[((size_t)b * S + (i0 + wv)) * S + j0 + lane] = sum * (1.0f / H);
    }

    // this head's P in A-fragment layout
    const half_t* prow = &pt[wv][c * 32];
    const v16h ap = cat8(*(const v8h*)(prow + hh * 16),
                         *(const v8h*)(prow + 8 + hh * 16));

    wait_async0();     // V chunk resident in vbuf[wv]
#pragma unroll
    for (int nt = 0; nt < 4; ++nt) {
      v16h bv;
#pragma unroll
      for (int i = 0; i < 16; ++i)
        bv[i] = vbuf[wv][(hh * 16 + i) * 64 + nt * 16 + c];
      acc[nt] = wmma_f16(ap, bv, acc[nt]);
    }
    __syncthreads();   // mean reads done before next chunk overwrites pt
  }

  // coalesced f16 store of this head's 16x64 attention-output tile
#pragma unroll
  for (int nt = 0; nt < 4; ++nt)
#pragma unroll
    for (int v = 0; v < 8; ++v)
      otile[wv][(v + 8 * hh) * 64 + nt * 16 + c] = (half_t)acc[nt][v];
  wait_ds0();          // wave-local tile ready (LDS in-order per wave)

  const int r2  = lane & 15;
  const int seg = lane >> 4;
  half_t* dst = ows + ((size_t)b * S + (i0 + r2)) * E + h * D + seg * 32;
  const half_t* src = &otile[wv][r2 * 64 + seg * 32];
#pragma unroll
  for (int t = 0; t < 4; ++t)
    *(v4u*)(dst + t * 8) = *(const v4u*)(src + t * 8);
}

// ---------------------------------------------------------------------------
// Kernel 3: output projection, 16x64 tile per wave, coalesced f32 epilogue.
// grid = (S/16, E/64, B)
// ---------------------------------------------------------------------------
__global__ void __launch_bounds__(32)
out_proj_kernel(const half_t* __restrict__ ows, const float* __restrict__ ow,
                const float* __restrict__ ob, float* __restrict__ out)
{
  __shared__ __align__(16) float ftile[16 * 64];

  const int lane = threadIdx.x & 31;
  const int c  = lane & 15;
  const int hh = lane >> 4;
  const int s0 = blockIdx.x * 16;
  const int n0 = blockIdx.y * 64;
  const int b  = blockIdx.z;

  const half_t* arow = ows + ((size_t)b * S + (s0 + c)) * E;
  const float*  wcol = ow + (size_t)(n0 + c) * E;

  v8f acc[4] = {};
  for (int kk = 0; kk < E; kk += 32) {
    const v16h a = cat8(*(const v8h*)(arow + kk + hh * 16),
                        *(const v8h*)(arow + kk + 8 + hh * 16));
#pragma unroll
    for (int nt = 0; nt < 4; ++nt) {
      const float* wr = wcol + (size_t)nt * 16 * E + kk + hh * 16;
      const v8f wb0 = *(const v8f*)wr;
      const v8f wb1 = *(const v8f*)(wr + 8);
      v16h bf;
#pragma unroll
      for (int i = 0; i < 8; ++i) { bf[i] = (half_t)wb0[i]; bf[i + 8] = (half_t)wb1[i]; }
      acc[nt] = wmma_f16(a, bf, acc[nt]);
    }
  }

#pragma unroll
  for (int nt = 0; nt < 4; ++nt)
#pragma unroll
    for (int v = 0; v < 8; ++v) {
      const int n = n0 + nt * 16 + c;
      ftile[(v + 8 * hh) * 64 + nt * 16 + c] = acc[nt][v] + ob[n];
    }
  __syncthreads();

  const int r2  = lane & 15;
  const int seg = lane >> 4;
  float* dst = out + ((size_t)(s0 + r2) * B + b) * E + n0 + seg * 32;
  const float* src = &ftile[r2 * 64 + seg * 32];
#pragma unroll
  for (int t = 0; t < 8; ++t)
    *(v4u*)(dst + t * 4) = *(const v4u*)(src + t * 4);
}

// ---------------------------------------------------------------------------
extern "C" void kernel_launch(void* const* d_in, const int* in_sizes, int n_in,
                              void* d_out, int out_size, void* d_ws, size_t ws_size,
                              hipStream_t stream)
{
  const float* x            = (const float*)d_in[0];
  const unsigned char* kpm  = (const unsigned char*)d_in[1];  // bool (1B)
  const unsigned char* cond = (const unsigned char*)d_in[2];  // bool (1B)
  const float* qw = (const float*)d_in[3];
  const float* qb = (const float*)d_in[4];
  const float* kw = (const float*)d_in[5];
  const float* kb = (const float*)d_in[6];
  const float* vw = (const float*)d_in[7];
  const float* vb = (const float*)d_in[8];
  const float* ow = (const float*)d_in[9];
  const float* ob = (const float*)d_in[10];

  float* out       = (float*)d_out;                 // S*B*E
  float* attn_mean = out + (size_t)S * B * E;       // B*S*S (fully stored)

  half_t* qws = (half_t*)d_ws;                      // f16 [B][H][S][D]
  half_t* kws = qws + (size_t)B * H * S * D;
  half_t* vws = kws + (size_t)B * H * S * D;
  half_t* ows = vws + (size_t)B * H * S * D;        // f16 [B][S][E]

  dim3 gproj(S / 16, E / 64, B * 3);
  qkv_proj_kernel<<<gproj, 32, 0, stream>>>(x, qw, qb, kw, kb, vw, vb,
                                            qws, kws, vws);

  dim3 gattn(S / 16, B);
  attention_kernel<<<gattn, 512, 0, stream>>>(qws, kws, vws, cond, kpm,
                                              attn_mean, ows);

  dim3 gout(S / 16, E / 64, B);
  out_proj_kernel<<<gout, 32, 0, stream>>>(ows, ow, ob, out);
}
